// GraphAttentionLayer_87308095193806
// MI455X (gfx1250) — matrix operand order repaired and checked by reading, not compiled, not run on hardware
//
#include <hip/hip_runtime.h>
#include <hip/hip_bf16.h>

#define N_NODES 8192
#define N_IN    512
#define N_OUT   256
#define ALPHA   0.2f
#define NEG_BIG -9.0e15f

typedef __attribute__((ext_vector_type(16))) _Float16 v16h;
typedef __attribute__((ext_vector_type(8)))  float    v8f;

static __device__ __forceinline__ v8f wmma_f16(v16h a, v16h b, v8f c) {
  // D = A(16x32 f16) * B(32x16 f16) + C(16x16 f32)
  return __builtin_amdgcn_wmma_f32_16x16x32_f16(
      /*neg_a=*/false, a, /*neg_b=*/false, b,
      /*c_mod=*/(short)0, c, /*reuse_a=*/false, /*reuse_b=*/false);
}

// ---------------------------------------------------------------------------
// k0: W (512x256 f32, row-major) -> Wt (256x512 f16)  [B-operand friendly]
// ---------------------------------------------------------------------------
__global__ void __launch_bounds__(256)
k_w_transpose(const float* __restrict__ W, _Float16* __restrict__ Wt) {
  int idx = blockIdx.x * 256 + threadIdx.x;   // 0 .. 512*256-1
  int k = idx >> 8;                           // / 256
  int n = idx & 255;
  Wt[n * N_IN + k] = (_Float16)W[idx];
}

// ---------------------------------------------------------------------------
// k1: h_W = h @ W  (8192x512 @ 512x256) via WMMA f16, f32 accumulate.
// Block = 128 thr = 4 waves; block owns 16 rows x 256 cols; wave owns 4 tiles.
// Stores h_W (f32 row-major) and h_Wt (f16, transposed [256][8192]).
// ---------------------------------------------------------------------------
__global__ void __launch_bounds__(128)
k_gemm_hw(const float* __restrict__ h, const _Float16* __restrict__ Wt,
          float* __restrict__ hW, _Float16* __restrict__ hWt) {
  const int lane = threadIdx.x & 31;
  const int wave = threadIdx.x >> 5;          // 0..3
  const int rowBase = blockIdx.x * 16;
  const int rsub = lane & 15;
  const int half = lane >> 4;                 // lane-half selects K sub-blocks
  const int grow = rowBase + rsub;

  v8f acc[4] = {};
  const float* hrow = h + (size_t)grow * N_IN;

  for (int kk = 0; kk < N_IN; kk += 32) {
    // A tile 16x32 f16: lanes 0-15 hold K 0..15, lanes 16-31 K 16..31
    float4 a0 = *(const float4*)(hrow + kk + half * 16);
    float4 a1 = *(const float4*)(hrow + kk + half * 16 + 4);
    float4 a2 = *(const float4*)(hrow + kk + half * 16 + 8);
    float4 a3 = *(const float4*)(hrow + kk + half * 16 + 12);
    v16h A;
    A[0]  = (_Float16)a0.x; A[1]  = (_Float16)a0.y; A[2]  = (_Float16)a0.z; A[3]  = (_Float16)a0.w;
    A[4]  = (_Float16)a1.x; A[5]  = (_Float16)a1.y; A[6]  = (_Float16)a1.z; A[7]  = (_Float16)a1.w;
    A[8]  = (_Float16)a2.x; A[9]  = (_Float16)a2.y; A[10] = (_Float16)a2.z; A[11] = (_Float16)a2.w;
    A[12] = (_Float16)a3.x; A[13] = (_Float16)a3.y; A[14] = (_Float16)a3.z; A[15] = (_Float16)a3.w;

#pragma unroll
    for (int t = 0; t < 4; ++t) {
      const int col = (wave * 4 + t) * 16 + rsub;
      // B 32x16: lane-half selects K 0-15 vs 16-31, contiguous in Wt
      v16h B = *(const v16h*)(Wt + (size_t)col * N_IN + kk + half * 16);
      acc[t] = wmma_f16(A, B, acc[t]);
    }
  }

  // C/D layout: VGPR r -> M = r + 8*half ; N = lane&15 (within tile)
#pragma unroll
  for (int t = 0; t < 4; ++t) {
    const int col = (wave * 4 + t) * 16 + rsub;
#pragma unroll
    for (int r = 0; r < 8; ++r) {
      const int m = r + half * 8;
      const float v = acc[t][r];
      hW[(size_t)(rowBase + m) * N_OUT + col] = v;
      hWt[(size_t)col * N_NODES + rowBase + m] = (_Float16)v;
    }
  }
}

// ---------------------------------------------------------------------------
// k2: f1[i] = h_W[i,:] . a1 ; f2[i] = h_W[i,:] . a2  (wave per row)
// ---------------------------------------------------------------------------
__global__ void __launch_bounds__(256)
k_attn_vec(const float* __restrict__ hW, const float* __restrict__ a,
           float* __restrict__ f1, float* __restrict__ f2) {
  const int lane = threadIdx.x & 31;
  const int wave = threadIdx.x >> 5;
  const int row = blockIdx.x * 8 + wave;
  const float* hr = hW + (size_t)row * N_OUT;
  float s1 = 0.f, s2 = 0.f;
#pragma unroll
  for (int i = 0; i < 2; ++i) {
    const int c = lane * 8 + i * 4;
    float4 hv  = *(const float4*)(hr + c);
    float4 av1 = *(const float4*)(a + c);
    float4 av2 = *(const float4*)(a + N_OUT + c);
    s1 += hv.x * av1.x + hv.y * av1.y + hv.z * av1.z + hv.w * av1.w;
    s2 += hv.x * av2.x + hv.y * av2.y + hv.z * av2.z + hv.w * av2.w;
  }
#pragma unroll
  for (int off = 16; off; off >>= 1) {
    s1 += __shfl_xor(s1, off, 32);
    s2 += __shfl_xor(s2, off, 32);
  }
  if (lane == 0) { f1[row] = s1; f2[row] = s2; }
}

// ---------------------------------------------------------------------------
// k3: FUSED flash-style masked softmax + (attention @ h_W) + ELU.
// One wave per 16-row block (32-thr workgroups, 512 blocks -> spreads across
// WGPs for HBM bandwidth). adj is read exactly ONCE (256 MB total).
//
// Per 32-wide j-chunk:
//   - lane (l) owns row l&15; lane pair (l, l+16) holds disjoint K-halves
//   - online max/sum per row, combined across the pair via shfl_xor(16)
//   - unnormalized probs exp(val - m_running) <= 1 -> f16 A-tile
//   - accumulators rescaled only when the running max moves (ballot-guarded,
//     wave-uniform branch; EXEC stays all-1s at every WMMA)
//   - 16 WMMAs accumulate the full 256-col output strip
// Epilogue folds 1/s and ELU.
// ---------------------------------------------------------------------------
__global__ void __launch_bounds__(32)
k_attn_apply(const int* __restrict__ adj, const float* __restrict__ f1,
             const float* __restrict__ f2, const _Float16* __restrict__ hWt,
             float* __restrict__ out) {
  const int lane = threadIdx.x & 31;
  const int rowBase = blockIdx.x * 16;
  const int rsub = lane & 15;
  const int half = lane >> 4;
  const int row = rowBase + rsub;                    // A-operand row for this lane

  const float f1r = f1[row];
  const int* arow = adj + (size_t)row * N_NODES;

  v8f acc[16] = {};
  float m = -__builtin_inff();                       // running row max (pair-identical)
  float s = 0.f;                                     // running row sum  (pair-identical)

  for (int jc = 0; jc < N_NODES; jc += 32) {
    // ---- A-side raw scores: lane's K indices {half*16+0..15}
    const int j0 = jc + half * 16;
    const int j1 = jc + half * 16 + 8;
    int4 av0 = *(const int4*)(arow + j0);
    int4 av1 = *(const int4*)(arow + j0 + 4);
    int4 av2 = *(const int4*)(arow + j1);
    int4 av3 = *(const int4*)(arow + j1 + 4);
    float4 fv0 = *(const float4*)(f2 + j0);
    float4 fv1 = *(const float4*)(f2 + j0 + 4);
    float4 fv2 = *(const float4*)(f2 + j1);
    float4 fv3 = *(const float4*)(f2 + j1 + 4);
    __builtin_prefetch(arow + jc + 2048, 0, 3);      // speculative stream-ahead

    auto raw = [&](int am, float fj) -> float {
      float e = f1r + fj;
      e = e > 0.f ? e : ALPHA * e;                   // LeakyReLU
      return am > 0 ? e : NEG_BIG;                   // mask
    };
    float v[16];
    v[0]  = raw(av0.x, fv0.x); v[1]  = raw(av0.y, fv0.y);
    v[2]  = raw(av0.z, fv0.z); v[3]  = raw(av0.w, fv0.w);
    v[4]  = raw(av1.x, fv1.x); v[5]  = raw(av1.y, fv1.y);
    v[6]  = raw(av1.z, fv1.z); v[7]  = raw(av1.w, fv1.w);
    v[8]  = raw(av2.x, fv2.x); v[9]  = raw(av2.y, fv2.y);
    v[10] = raw(av2.z, fv2.z); v[11] = raw(av2.w, fv2.w);
    v[12] = raw(av3.x, fv3.x); v[13] = raw(av3.y, fv3.y);
    v[14] = raw(av3.z, fv3.z); v[15] = raw(av3.w, fv3.w);

    // ---- online softmax update (row = lane pair)
    float cmax = v[0];
#pragma unroll
    for (int i = 1; i < 16; ++i) cmax = fmaxf(cmax, v[i]);
    cmax = fmaxf(cmax, __shfl_xor(cmax, 16, 32));    // combine lane pair
    const float mn = fmaxf(m, cmax);
    const float scale = __expf(m - mn);              // first iter: exp(-inf)=0
    const bool upd = mn > m;
    m = mn;

    v16h A;
    float psum = 0.f;
#pragma unroll
    for (int i = 0; i < 16; ++i) {
      const float p = __expf(v[i] - mn);             // <= 1, masked -> 0
      psum += p;
      A[i] = (_Float16)p;
    }
    psum += __shfl_xor(psum, 16, 32);
    s = s * scale + psum;

    // ---- rare accumulator rescale (wave-uniform branch)
    if (__ballot(upd) != 0ull) {
#pragma unroll
      for (int r = 0; r < 8; ++r) {
        const float sc = __shfl(scale, r + 8 * half, 32);  // scale of row M=r+8*half
#pragma unroll
        for (int t = 0; t < 16; ++t) acc[t][r] *= sc;
      }
    }

    // ---- 16 WMMAs: full 256-col output strip
#pragma unroll
    for (int t = 0; t < 16; ++t) {
      const int col = t * 16 + rsub;
      v16h B = *(const v16h*)(hWt + (size_t)col * N_NODES + jc + half * 16);
      acc[t] = wmma_f16(A, B, acc[t]);
    }
  }

  // ---- epilogue: fold 1/s, ELU(alpha=1), store f32
  const float rs = 1.0f / s;
#pragma unroll
  for (int r = 0; r < 8; ++r) {
    const float sc = __shfl(rs, r + 8 * half, 32);   // 1/s of row M=r+8*half
    const int gm = rowBase + r + half * 8;
#pragma unroll
    for (int t = 0; t < 16; ++t) {
      const int col = t * 16 + rsub;
      float x = acc[t][r] * sc;
      x = x > 0.f ? x : (__expf(x) - 1.0f);
      out[(size_t)gm * N_OUT + col] = x;
    }
  }
}

// ---------------------------------------------------------------------------
extern "C" void kernel_launch(void* const* d_in, const int* in_sizes, int n_in,
                              void* d_out, int out_size, void* d_ws, size_t ws_size,
                              hipStream_t stream) {
  (void)in_sizes; (void)n_in; (void)out_size; (void)ws_size;
  const float* h   = (const float*)d_in[0];   // (8192, 512) f32
  const int*   adj = (const int*)d_in[1];     // (8192, 8192) i32
  const float* W   = (const float*)d_in[2];   // (512, 256) f32
  const float* a   = (const float*)d_in[3];   // (512, 1) f32
  float* out = (float*)d_out;                 // (8192, 256) f32

  char* ws = (char*)d_ws;
  float*    hW  = (float*)ws;     ws += (size_t)N_NODES * N_OUT * sizeof(float);
  _Float16* hWt = (_Float16*)ws;  ws += (size_t)N_NODES * N_OUT * sizeof(_Float16);
  _Float16* Wt  = (_Float16*)ws;  ws += (size_t)N_IN * N_OUT * sizeof(_Float16);
  float* f1 = (float*)ws;  ws += N_NODES * sizeof(float);
  float* f2 = (float*)ws;  ws += N_NODES * sizeof(float);

  k_w_transpose<<<(N_IN * N_OUT) / 256, 256, 0, stream>>>(W, Wt);
  k_gemm_hw    <<<N_NODES / 16,        128, 0, stream>>>(h, Wt, hW, hWt);
  k_attn_vec   <<<N_NODES / 8,         256, 0, stream>>>(hW, a, f1, f2);
  k_attn_apply <<<N_NODES / 16,         32, 0, stream>>>(adj, f1, f2, hWt, out);
}
